// StandaloneMLA_59614146068720
// MI455X (gfx1250) — compile-verified, matrix operand order checked by reading, not run on hardware
//
#include <hip/hip_runtime.h>
#include <hip/hip_bf16.h>

// ---- problem constants (from reference) ----
#define B_    2
#define S_    2048
#define HID_  2048
#define H_    16
#define NOPE_ 128
#define ROPE_ 64
#define QKD_  192      // NOPE + ROPE
#define VD_   128
#define QR_   1536
#define KVR_  512
#define BS_   (B_*S_)

typedef __bf16 bf16;
typedef __attribute__((ext_vector_type(16))) __bf16 v16bf;
typedef __attribute__((ext_vector_type(8)))  __bf16 v8bf;
typedef __attribute__((ext_vector_type(8)))  float  v8f;
typedef __attribute__((ext_vector_type(4)))  unsigned v4u;
typedef __attribute__((ext_vector_type(8)))  unsigned v8u;

// ---------------- CDNA5 async global->LDS copy (ASYNCcnt) ---------------
__device__ __forceinline__ void async_ld_b128(bf16* lds_dst, const bf16* gsrc) {
  unsigned off = (unsigned)(size_t)lds_dst;   // low 32 bits == LDS offset
  asm volatile("global_load_async_to_lds_b128 %0, %1, off"
               :: "v"(off), "v"(gsrc) : "memory");
}
__device__ __forceinline__ void wait_async0() {
  asm volatile("s_wait_asynccnt 0x0" ::: "memory");
}
__device__ __forceinline__ void wait_ds0() {
  asm volatile("s_wait_dscnt 0x0" ::: "memory");
}
__device__ __forceinline__ void wait_tensor0() {
  __builtin_amdgcn_s_wait_tensorcnt(0);       // probe-confirmed builtin
}

// ---------------- CDNA5 Tensor Data Mover 2D tile load ------------------
// Builds D# group0 (128b) + group1 (256b) per ISA 08_async_tensor §8 and
// issues tensor_load_to_lds (2-group form, <=2D tensors). Units are 8 bytes
// (data_size=3). LDS padding: pad_amt DWORD-code inserted every
// pad_int DWORD-code of data (hardware bank-conflict padding).
__device__ __forceinline__ void tdm_load_2d(unsigned lds_off, const void* gaddr,
                                            unsigned tile_d0_u8, unsigned tile_d1,
                                            unsigned row_stride_u8,
                                            unsigned pad_int_code, unsigned pad_amt_code) {
  unsigned long long ga = (unsigned long long)(size_t)gaddr;
  v4u g0;
  g0.x = 1u;                                   // count=1 (valid user descriptor)
  g0.y = lds_off;                              // lds_addr (bytes)
  g0.z = (unsigned)ga;                         // global_addr[31:0]
  g0.w = ((unsigned)(ga >> 32) & 0x01ffffffu) | 0x80000000u;  // addr[56:32], type=2
  v8u g1;
  g1.s0 = (3u << 16) | (1u << 20)              // data_size=8B, pad_enable
        | (pad_int_code << 22) | (pad_amt_code << 25);
  g1.s1 = 0u;                                  // abar=0; tensor_dim0[15:0] (dim0 = 1<<30)
  g1.s2 = 0x4000u;                             // tensor_dim0[31:16]; tensor_dim1[15:0]
  g1.s3 = 0x4000u | (tile_d0_u8 << 16);        // tensor_dim1[31:16]=2^30; tile_dim0
  g1.s4 = tile_d1 & 0xffffu;                   // tile_dim1 (tile_dim2 = 0)
  g1.s5 = row_stride_u8;                       // tensor_dim0_stride[31:0]
  g1.s6 = 0u;                                  // stride hi / dim1_stride lo
  g1.s7 = 0u;                                  // dim1_stride hi
  asm volatile("tensor_load_to_lds %0, %1" :: "s"(g0), "s"(g1) : "memory");
}

// Load a 16x32 bf16 WMMA operand fragment (A-layout; same per-lane pattern
// serves as B-layout when the source holds the transposed matrix row-major).
__device__ __forceinline__ v16bf load_frag(const bf16* __restrict__ base, int ld, int row0, int k0) {
  const int lane = threadIdx.x & 31;
  const bf16* p = base + (size_t)(row0 + (lane & 15)) * ld + k0 + ((lane >> 4) << 3);
  union { v16bf v; v8bf h[2]; } u;
  u.h[0] = *(const v8bf*)(p);
  u.h[1] = *(const v8bf*)(p + 16);
  return u.v;
}

// ---------------- bf16 WMMA GEMM:  C[M,N] = A[M,K] * W[N,K]^T -----------
// block = 128 threads (4 waves), block tile 64x64, wave tile 32x32,
// K-step 64, TDM double-buffered LDS staging (wave 0 drives the DMA).
#define GLD 72   // LDS row: 64 data + 8 pad bf16 (144B -> 36 banks, conflict-free)
template<bool F32OUT>
__global__ __launch_bounds__(128)
void gemm_nt(const bf16* __restrict__ A, const bf16* __restrict__ W,
             void* __restrict__ Cout, int M, int N, int K) {
  __shared__ __align__(16) bf16 At[2][64 * GLD];
  __shared__ __align__(16) bf16 Wt[2][64 * GLD];

  const int wid  = threadIdx.x >> 5;
  const int lane = threadIdx.x & 31;
  const int mB = blockIdx.y * 64, nB = blockIdx.x * 64;
  const int m0 = (wid >> 1) * 32, n0 = (wid & 1) * 32;

  // one 64x64 bf16 tile each for A and W: row = 128B (16 x 8B units),
  // pad 16B after every 128B (pad_int code 4 = 32 DW, pad_amt code 3 = 4 DW)
  auto stage = [&](int buf, int k0) {
    if (wid == 0) {
      tdm_load_2d((unsigned)(size_t)&At[buf][0], A + (size_t)mB * K + k0,
                  16, 64, (unsigned)(K >> 2), 4, 3);
      tdm_load_2d((unsigned)(size_t)&Wt[buf][0], W + (size_t)nB * K + k0,
                  16, 64, (unsigned)(K >> 2), 4, 3);
    }
  };

  stage(0, 0);
  if (wid == 0) wait_tensor0();
  __syncthreads();

  v8f acc[2][2] = {};
  for (int k0 = 0; k0 < K; k0 += 64) {
    const int buf = (k0 >> 6) & 1;
    if (k0 + 64 < K) stage(buf ^ 1, k0 + 64);
#pragma unroll
    for (int kk = 0; kk < 2; ++kk) {
      v16bf a0 = load_frag(At[buf], GLD, m0,      kk * 32);
      v16bf a1 = load_frag(At[buf], GLD, m0 + 16, kk * 32);
      v16bf b0 = load_frag(Wt[buf], GLD, n0,      kk * 32);
      v16bf b1 = load_frag(Wt[buf], GLD, n0 + 16, kk * 32);
      acc[0][0] = __builtin_amdgcn_wmma_f32_16x16x32_bf16(false, a0, false, b0, (short)0, acc[0][0], false, false);
      acc[0][1] = __builtin_amdgcn_wmma_f32_16x16x32_bf16(false, a0, false, b1, (short)0, acc[0][1], false, false);
      acc[1][0] = __builtin_amdgcn_wmma_f32_16x16x32_bf16(false, a1, false, b0, (short)0, acc[1][0], false, false);
      acc[1][1] = __builtin_amdgcn_wmma_f32_16x16x32_bf16(false, a1, false, b1, (short)0, acc[1][1], false, false);
    }
    if (wid == 0) wait_tensor0();   // next tiles landed
    __syncthreads();                // everyone done reading current tiles
  }

  const int g = lane >> 4, n = lane & 15;
#pragma unroll
  for (int i = 0; i < 2; ++i)
#pragma unroll
    for (int j = 0; j < 2; ++j) {
      const int mb  = mB + m0 + i * 16 + g * 8;
      const int col = nB + n0 + j * 16 + n;
#pragma unroll
      for (int r = 0; r < 8; ++r) {
        if constexpr (F32OUT) ((float*)Cout)[(size_t)(mb + r) * N + col] = acc[i][j][r];
        else                  ((bf16*)Cout)[(size_t)(mb + r) * N + col] = (bf16)acc[i][j][r];
      }
    }
}

// ---------------- fp32 -> bf16 convert ----------------
__global__ void cvt_f32_bf16(const float* __restrict__ in, bf16* __restrict__ out, size_t n) {
  size_t i = (size_t)blockIdx.x * blockDim.x + threadIdx.x;
  if (i < n) out[i] = (bf16)in[i];
}

// ---------------- RMS norm: fp32 in (row stride in_ld), bf16 out --------
__global__ __launch_bounds__(256)
void rmsnorm(const float* __restrict__ in, int in_ld, int C,
             const float* __restrict__ w, bf16* __restrict__ out, int out_ld) {
  const int row = blockIdx.x;
  const float* x = in + (size_t)row * in_ld;
  float ss = 0.f;
  for (int i = threadIdx.x; i < C; i += blockDim.x) { float v = x[i]; ss += v * v; }
#pragma unroll
  for (int m = 1; m < 32; m <<= 1) ss += __shfl_xor(ss, m, 32);
  __shared__ float part[8];
  __shared__ float s_inv;
  if ((threadIdx.x & 31) == 0) part[threadIdx.x >> 5] = ss;
  __syncthreads();
  if (threadIdx.x == 0) {
    float t = 0.f;
    for (int i = 0; i < 8; ++i) t += part[i];
    s_inv = rsqrtf(t / (float)C + 1e-6f);
  }
  __syncthreads();
  const float inv = s_inv;
  bf16* y = out + (size_t)row * out_ld;
  for (int i = threadIdx.x; i < C; i += blockDim.x) y[i] = (bf16)(x[i] * inv * w[i]);
}

// ---------------- RoPE on q_pe; write head-major q (B,H,S,QKD) ----------
__global__ void rope_q(const bf16* __restrict__ qraw, const int* __restrict__ pid,
                       bf16* __restrict__ qout) {
  const int bs = blockIdx.x;
  const int b = bs / S_, s = bs % S_;
  const float pos = (float)pid[bs];
  const bf16* src = qraw + (size_t)bs * (H_ * QKD_);
  for (int e = threadIdx.x; e < H_ * NOPE_; e += blockDim.x) {
    int h = e >> 7, d = e & 127;
    qout[(((size_t)(b * H_ + h)) * S_ + s) * QKD_ + d] = src[h * QKD_ + d];
  }
  for (int p = threadIdx.x; p < H_ * 32; p += blockDim.x) {
    int h = p >> 5, i = p & 31;
    float x0 = (float)src[h * QKD_ + NOPE_ + 2 * i];
    float x1 = (float)src[h * QKD_ + NOPE_ + 2 * i + 1];
    float fr = __powf(10000.f, -(float)i / 32.f);
    float cs = __cosf(pos * fr), sn = __sinf(pos * fr);
    bf16* dst = qout + (((size_t)(b * H_ + h)) * S_ + s) * QKD_ + NOPE_;
    dst[i]      = (bf16)(x0 * cs - x1 * sn);   // deinterleave + rotate_half
    dst[32 + i] = (bf16)(x1 * cs + x0 * sn);
  }
}

// ------- arrange kv: k (B,H,S,QKD) with RoPE'd broadcast k_pe; vT (B,H,VD,S)
__global__ void arrange_kv(const bf16* __restrict__ kvraw, const float* __restrict__ ckv,
                           const int* __restrict__ pid, bf16* __restrict__ kout,
                           bf16* __restrict__ vtout) {
  const int bs = blockIdx.x;
  const int b = bs / S_, s = bs % S_;
  const bf16* src = kvraw + (size_t)bs * (H_ * (NOPE_ + VD_));
  for (int e = threadIdx.x; e < H_ * (NOPE_ + VD_); e += blockDim.x) {
    int h = e >> 8, d = e & 255;
    if (d < NOPE_)
      kout[(((size_t)(b * H_ + h)) * S_ + s) * QKD_ + d] = src[h * 256 + d];
    else
      vtout[(((size_t)(b * H_ + h)) * VD_ + (d - NOPE_)) * S_ + s] = src[h * 256 + d];
  }
  const float pos = (float)pid[bs];
  const float* pe = ckv + (size_t)bs * (KVR_ + ROPE_) + KVR_;
  for (int i = threadIdx.x; i < 32; i += blockDim.x) {
    float x0 = pe[2 * i], x1 = pe[2 * i + 1];
    float fr = __powf(10000.f, -(float)i / 32.f);
    float cs = __cosf(pos * fr), sn = __sinf(pos * fr);
    float o0 = x0 * cs - x1 * sn, o1 = x1 * cs + x0 * sn;
    for (int h = 0; h < H_; ++h) {
      bf16* dst = kout + (((size_t)(b * H_ + h)) * S_ + s) * QKD_ + NOPE_;
      dst[i] = (bf16)o0; dst[32 + i] = (bf16)o1;
    }
  }
}

// ---------------- causal flash attention, WMMA bf16 ---------------------
// grid: (S/64, B*H). block 128 = 4 waves; wave owns 16 query rows.
// Key block = 64. K tile via async-to-LDS (ASYNCcnt); V^T tile via TDM
// (TENSORcnt, wave 0). Double-buffered, shared by all 4 waves.
#define KLD 200   // 192 + 8 pad bf16
#define VLD 72    // 64 + 8 pad bf16 (matches TDM pad: 128B data + 16B pad)
#define PLD 72    // 64 + 8 pad bf16
__global__ __launch_bounds__(128)
void mla_attn(const bf16* __restrict__ Q, const bf16* __restrict__ Kf,
              const bf16* __restrict__ VT, bf16* __restrict__ Out) {
  __shared__ __align__(16) bf16 Kt[2][64 * KLD];
  __shared__ __align__(16) bf16 Vt[2][128 * VLD];
  __shared__ __align__(16) bf16 Pl[4][16 * PLD];

  const int bh = blockIdx.y;
  const int b = bh >> 4, h = bh & 15;
  const int wid = threadIdx.x >> 5, lane = threadIdx.x & 31;
  const int q0 = blockIdx.x * 64 + wid * 16;
  const bf16* qbase = Q  + (size_t)bh * S_ * QKD_;
  const bf16* kbase = Kf + (size_t)bh * S_ * QKD_;
  const bf16* vbase = VT + (size_t)bh * VD_ * S_;

  // stage 64 keys: K rows (64 x 192, async) and V^T tile (128 x 64, TDM)
  auto stage = [&](int buf, int kstart) {
#pragma unroll
    for (int rep = 0; rep < 12; ++rep) {          // 64*24 chunks / 128 thr
      int c = threadIdx.x + rep * 128;
      int row = c / 24, j = c % 24;
      async_ld_b128(&Kt[buf][row * KLD + j * 8],
                    kbase + (size_t)(kstart + row) * QKD_ + j * 8);
    }
    if (wid == 0)                                 // 128 rows x 128B, stride S*2
      tdm_load_2d((unsigned)(size_t)&Vt[buf][0], vbase + kstart,
                  16, 128, (unsigned)(S_ >> 2), 4, 3);
  };

  v16bf qf[6];
#pragma unroll
  for (int c = 0; c < 6; ++c) qf[c] = load_frag(qbase, QKD_, q0, c * 32);

  v8f o[8] = {};
  float mrow[8], lrow[8];
#pragma unroll
  for (int r = 0; r < 8; ++r) { mrow[r] = -__builtin_inff(); lrow[r] = 0.f; }

  const int g = lane >> 4, nn = lane & 15;
  const float scale = 0.07216878364870323f;       // QKD^-0.5

  const int nkb = blockIdx.x + 1;                 // causal: keys <= qblk+63
  stage(0, 0);
  wait_async0();
  if (wid == 0) wait_tensor0();
  __syncthreads();

  for (int kb = 0; kb < nkb; ++kb) {
    const int buf = kb & 1;
    const int kstart = kb * 64;
    if (kb + 1 < nkb) stage(buf ^ 1, kstart + 64);

    // S = Q K^T over 64 keys (4 sub-tiles of 16)
    v8f sacc[4] = {};
#pragma unroll
    for (int c = 0; c < 6; ++c) {
#pragma unroll
      for (int t = 0; t < 4; ++t) {
        v16bf kf = load_frag(Kt[buf], KLD, t * 16, c * 32);
        sacc[t] = __builtin_amdgcn_wmma_f32_16x16x32_bf16(false, qf[c], false, kf, (short)0, sacc[t], false, false);
      }
    }
    // mask + scale + online softmax (row stats uniform per 16-lane half)
#pragma unroll
    for (int r = 0; r < 8; ++r) {
      const int qrow = q0 + g * 8 + r;
      float sv[4], v = -__builtin_inff();
#pragma unroll
      for (int t = 0; t < 4; ++t) {
        float s = sacc[t][r] * scale;
        if (kstart + t * 16 + nn > qrow) s = -__builtin_inff();
        sv[t] = s;
        v = fmaxf(v, s);
      }
#pragma unroll
      for (int msk = 1; msk < 16; msk <<= 1) v = fmaxf(v, __shfl_xor(v, msk, 32));
      const float nm = fmaxf(mrow[r], v);
      const float f = __expf(mrow[r] - nm);
      mrow[r] = nm;
      lrow[r] *= f;
#pragma unroll
      for (int c = 0; c < 8; ++c) o[c][r] *= f;
#pragma unroll
      for (int t = 0; t < 4; ++t) {
        const float p = __expf(sv[t] - nm);
        lrow[r] += p;                              // lane-local partial rowsum
        Pl[wid][(g * 8 + r) * PLD + t * 16 + nn] = (bf16)p;
      }
    }
    wait_ds0();
    // reload P in A-layout: two 16x32 fragments (keys 0..31, 32..63)
    union { v16bf v; v8bf x[2]; } pu[2];
    {
      const int prow = lane & 15;
      const int kh = (lane >> 4) << 3;
#pragma unroll
      for (int t2 = 0; t2 < 2; ++t2) {
        pu[t2].x[0] = *(const v8bf*)&Pl[wid][prow * PLD + t2 * 32 + kh];
        pu[t2].x[1] = *(const v8bf*)&Pl[wid][prow * PLD + t2 * 32 + 16 + kh];
      }
    }
    // O += P V  (V^T rows are output dims)
#pragma unroll
    for (int c = 0; c < 8; ++c) {
      v16bf vf0 = load_frag(Vt[buf], VLD, c * 16, 0);
      v16bf vf1 = load_frag(Vt[buf], VLD, c * 16, 32);
      o[c] = __builtin_amdgcn_wmma_f32_16x16x32_bf16(false, pu[0].v, false, vf0, (short)0, o[c], false, false);
      o[c] = __builtin_amdgcn_wmma_f32_16x16x32_bf16(false, pu[1].v, false, vf1, (short)0, o[c], false, false);
    }
    wait_async0();                   // next K tile landed (this wave's loads)
    if (wid == 0) wait_tensor0();    // next V tile landed
    __syncthreads();                 // all waves done with current tiles
  }

  // finalize: rowsum across 16-lane half, normalize, store (B,S,H*VD)
#pragma unroll
  for (int r = 0; r < 8; ++r) {
    float l = lrow[r];
#pragma unroll
    for (int msk = 1; msk < 16; msk <<= 1) l += __shfl_xor(l, msk, 32);
    lrow[r] = 1.0f / l;
  }
  bf16* obase = Out + (size_t)b * S_ * (H_ * VD_) + h * VD_;
#pragma unroll
  for (int c = 0; c < 8; ++c)
#pragma unroll
    for (int r = 0; r < 8; ++r) {
      const int qrow = q0 + g * 8 + r;
      obase[(size_t)qrow * (H_ * VD_) + c * 16 + nn] = (bf16)(o[c][r] * lrow[r]);
    }
}

// ======================= host launcher ==================================
extern "C" void kernel_launch(void* const* d_in, const int* in_sizes, int n_in,
                              void* d_out, int out_size, void* d_ws, size_t ws_size,
                              hipStream_t stream) {
  const float* hidden = (const float*)d_in[0];
  const int*   pid    = (const int*)d_in[1];
  const float* w_qa   = (const float*)d_in[2];
  const float* ln_qa  = (const float*)d_in[3];
  const float* w_qb   = (const float*)d_in[4];
  const float* w_kva  = (const float*)d_in[5];
  const float* ln_kva = (const float*)d_in[6];
  const float* w_kvb  = (const float*)d_in[7];
  const float* w_o    = (const float*)d_in[8];
  float* out = (float*)d_out;

  char* ws = (char*)d_ws;
  size_t off = 0;
  auto alloc = [&](size_t bytes) -> void* {
    void* p = ws + off;
    off = (off + bytes + 255) & ~(size_t)255;
    return p;
  };

  bf16*  hid_bf   = (bf16*)alloc((size_t)BS_ * HID_ * 2);
  bf16*  wqa_bf   = (bf16*)alloc((size_t)QR_ * HID_ * 2);
  bf16*  wqb_bf   = (bf16*)alloc((size_t)H_ * QKD_ * QR_ * 2);
  bf16*  wkva_bf  = (bf16*)alloc((size_t)(KVR_ + ROPE_) * HID_ * 2);
  bf16*  wkvb_bf  = (bf16*)alloc((size_t)H_ * (NOPE_ + VD_) * KVR_ * 2);
  bf16*  wo_bf    = (bf16*)alloc((size_t)HID_ * H_ * VD_ * 2);
  float* qa_f     = (float*)alloc((size_t)BS_ * QR_ * 4);
  float* ckv_f    = (float*)alloc((size_t)BS_ * (KVR_ + ROPE_) * 4);
  bf16*  qan_bf   = (bf16*)alloc((size_t)BS_ * QR_ * 2);
  bf16*  ckvn_bf  = (bf16*)alloc((size_t)BS_ * KVR_ * 2);
  bf16*  qraw_bf  = (bf16*)alloc((size_t)BS_ * H_ * QKD_ * 2);
  bf16*  kvraw_bf = (bf16*)alloc((size_t)BS_ * H_ * (NOPE_ + VD_) * 2);
  bf16*  q_bf     = (bf16*)alloc((size_t)B_ * H_ * S_ * QKD_ * 2);
  bf16*  k_bf     = (bf16*)alloc((size_t)B_ * H_ * S_ * QKD_ * 2);
  bf16*  vt_bf    = (bf16*)alloc((size_t)B_ * H_ * VD_ * S_ * 2);
  bf16*  ao_bf    = (bf16*)alloc((size_t)BS_ * H_ * VD_ * 2);

  auto cvt = [&](const float* src, bf16* dst, size_t n) {
    int blocks = (int)((n + 255) / 256);
    cvt_f32_bf16<<<blocks, 256, 0, stream>>>(src, dst, n);
  };
  cvt(hidden, hid_bf, (size_t)BS_ * HID_);
  cvt(w_qa,  wqa_bf,  (size_t)QR_ * HID_);
  cvt(w_qb,  wqb_bf,  (size_t)H_ * QKD_ * QR_);
  cvt(w_kva, wkva_bf, (size_t)(KVR_ + ROPE_) * HID_);
  cvt(w_kvb, wkvb_bf, (size_t)H_ * (NOPE_ + VD_) * KVR_);
  cvt(w_o,   wo_bf,   (size_t)HID_ * H_ * VD_);

  // q_a = hidden @ w_qa^T ; ckv = hidden @ w_kva^T (fp32 out for RMS norm)
  gemm_nt<true ><<<dim3(QR_ / 64, BS_ / 64), 128, 0, stream>>>(hid_bf, wqa_bf,  qa_f,  BS_, QR_, HID_);
  gemm_nt<true ><<<dim3((KVR_ + ROPE_) / 64, BS_ / 64), 128, 0, stream>>>(hid_bf, wkva_bf, ckv_f, BS_, KVR_ + ROPE_, HID_);
  rmsnorm<<<BS_, 256, 0, stream>>>(qa_f,  QR_,          QR_,  ln_qa,  qan_bf,  QR_);
  rmsnorm<<<BS_, 256, 0, stream>>>(ckv_f, KVR_ + ROPE_, KVR_, ln_kva, ckvn_bf, KVR_);
  // q = qa_norm @ w_qb^T ; kv = ckv_norm @ w_kvb^T (bf16 out)
  gemm_nt<false><<<dim3(H_ * QKD_ / 64, BS_ / 64), 128, 0, stream>>>(qan_bf,  wqb_bf,  qraw_bf,  BS_, H_ * QKD_, QR_);
  gemm_nt<false><<<dim3(H_ * (NOPE_ + VD_) / 64, BS_ / 64), 128, 0, stream>>>(ckvn_bf, wkvb_bf, kvraw_bf, BS_, H_ * (NOPE_ + VD_), KVR_);
  // RoPE + rearrangement
  rope_q<<<BS_, 256, 0, stream>>>(qraw_bf, pid, q_bf);
  arrange_kv<<<BS_, 256, 0, stream>>>(kvraw_bf, ckv_f, pid, k_bf, vt_bf);
  // causal attention
  mla_attn<<<dim3(S_ / 64, B_ * H_), 128, 0, stream>>>(q_bf, k_bf, vt_bf, ao_bf);
  // out = attn @ w_o^T (fp32)
  gemm_nt<true ><<<dim3(HID_ / 64, BS_ / 64), 128, 0, stream>>>(ao_bf, wo_bf, out, BS_, HID_, H_ * VD_);
}
